// GraphSAGE_62251255988403
// MI455X (gfx1250) — compile-verified, matrix operand order checked by reading, not run on hardware
//
#include <hip/hip_runtime.h>
#include <hip/hip_bf16.h>

typedef __attribute__((ext_vector_type(16))) __bf16 v16bf;
typedef __attribute__((ext_vector_type(8)))  float  v8f;

#define DH   128
#define DOUT 64

// WMMA bf16 A-fragment per-lane k order within a 32-wide chunk:
// lane group g = lane>>4 needs K = {g*8..g*8+7, 16+g*8..16+g*8+7}.
// We store activations so each lane fragment is contiguous:
// row layout per 32-chunk kc: [k=0..7][k=16..23][k=8..15][k=24..31]
__device__ __host__ inline int perm_off(int k) {
  int kc = k >> 5, kk = k & 31;
  int g, p;
  if (kk < 16) { g = kk >> 3; p = kk & 7; }
  else         { g = (kk - 16) >> 3; p = 8 + ((kk - 16) & 7); }
  return kc * 32 + g * 16 + p;
}

// ---------------------------------------------------------------- utilities
__global__ void zero_f32_kernel(float* __restrict__ p, long long n) {
  long long i = (long long)blockIdx.x * blockDim.x + threadIdx.x;
  if (i < n) p[i] = 0.0f;
}

// cnt[i] <- 1/max(cnt[i],1)
__global__ void invcnt_kernel(float* __restrict__ cnt, int n) {
  int i = blockIdx.x * blockDim.x + threadIdx.x;
  if (i < n) {
    float c = cnt[i];
    cnt[i] = 1.0f / (c > 1.0f ? c : 1.0f);
  }
}

// Weights -> bf16 in WMMA-packed layout: entry i = ((nt*4 + kc)*32 + lane)*16 + e
// lane element e -> K = kc*32 + (e<8 ? (lane>>4)*8+e : 16+(lane>>4)*8+e-8), N = nt*16+(lane&15)
// Also folds BN into per-channel scale/shift: relu(gemm*sc + sh)
__global__ void prep_layer_kernel(const float* __restrict__ Wl, const float* __restrict__ Wr,
                                  const float* __restrict__ bl, const float* __restrict__ g,
                                  const float* __restrict__ b,  const float* __restrict__ rm,
                                  const float* __restrict__ rv,
                                  __bf16* __restrict__ Wlp, __bf16* __restrict__ Wrp,
                                  float* __restrict__ sc, float* __restrict__ sh) {
  int i = blockIdx.x * blockDim.x + threadIdx.x;
  if (i < DH * DH) {
    int e = i & 15, lane = (i >> 4) & 31, kc = (i >> 9) & 3, nt = i >> 11; // nt 0..7
    int g8 = lane >> 4;
    int kk = (e < 8) ? (g8 * 8 + e) : (16 + g8 * 8 + (e - 8));
    int k = kc * 32 + kk;
    int n = nt * 16 + (lane & 15);
    Wlp[i] = (__bf16)Wl[k * DH + n];
    Wrp[i] = (__bf16)Wr[k * DH + n];
  }
  if (i < DH) {
    float s = g[i] * rsqrtf(rv[i] + 1e-5f);
    sc[i] = s;
    sh[i] = (bl[i] - rm[i]) * s + b[i];
  }
}

__global__ void prep_wo_kernel(const float* __restrict__ Wo, __bf16* __restrict__ Wop) {
  int i = blockIdx.x * blockDim.x + threadIdx.x;
  if (i < DH * DOUT) {
    int e = i & 15, lane = (i >> 4) & 31, kc = (i >> 9) & 3, nt = i >> 11; // nt 0..3
    int g8 = lane >> 4;
    int kk = (e < 8) ? (g8 * 8 + e) : (16 + g8 * 8 + (e - 8));
    int k = kc * 32 + kk;
    int n = nt * 16 + (lane & 15);
    Wop[i] = (__bf16)Wo[k * DOUT + n];
  }
}

// ------------------------------------------------------- edge scatter (L2 atomics)
// 128 consecutive lanes handle one edge -> fully coalesced gather + scatter.
__global__ void scatter_kernel(const float* __restrict__ h, const int* __restrict__ src,
                               const int* __restrict__ dst, float* __restrict__ agg,
                               float* __restrict__ cnt, int nE) {
  long long idx = (long long)blockIdx.x * blockDim.x + threadIdx.x;
  long long total = (long long)nE * DH;
  if (idx >= total) return;
  int e = (int)(idx >> 7);
  int d = (int)(idx & (DH - 1));
  int s = src[e];
  int t = dst[e];
  atomicAdd(&agg[(size_t)t * DH + d], h[(size_t)s * DH + d]);
  if (d == 0) atomicAdd(&cnt[t], 1.0f);
}

// mean = agg * cntinv -> bf16 (permuted layout); shadow-convert h (f32) -> bf16 (permuted)
__global__ void mean_conv_kernel(const float* __restrict__ agg, const float* __restrict__ cntinv,
                                 const float* __restrict__ hsrc, __bf16* __restrict__ meanb,
                                 __bf16* __restrict__ hb, int nrows) {
  long long idx = (long long)blockIdx.x * blockDim.x + threadIdx.x;
  if (idx >= (long long)nrows * DH) return;
  int row = (int)(idx >> 7);
  int k = (int)(idx & (DH - 1));
  long long dst = (long long)row * DH + perm_off(k);
  float inv = cntinv[row];
  meanb[dst] = (__bf16)(agg[idx] * inv);
  hb[dst]    = (__bf16)hsrc[idx];
}

// f32 -> bf16 with the same permuted layout (for the output head input)
__global__ void conv_perm_kernel(const float* __restrict__ in, __bf16* __restrict__ out,
                                 int nrows) {
  long long idx = (long long)blockIdx.x * blockDim.x + threadIdx.x;
  if (idx >= (long long)nrows * DH) return;
  int row = (int)(idx >> 7);
  int k = (int)(idx & (DH - 1));
  out[(long long)row * DH + perm_off(k)] = (__bf16)in[idx];
}

// ------------------------------ fused SAGE layer GEMM: mean@Wl + h@Wr, BN+ReLU
// 512 threads = 16 waves; wave(mt,nt) computes one 16x16 output tile.
// B fragments preloaded to registers (weights are L2-resident, 32KB each).
__global__ __launch_bounds__(512) void sage_gemm_kernel(
    const __bf16* __restrict__ meanb, const __bf16* __restrict__ hb,
    const __bf16* __restrict__ Wlp,   const __bf16* __restrict__ Wrp,
    const float* __restrict__ sc,     const float* __restrict__ sh,
    float* __restrict__ hout, int nrows) {
  int t = threadIdx.x;
  int wave = t >> 5, lane = t & 31;
  int mt = wave >> 3, nt = wave & 7;            // 2 x 8 tiles -> 32 rows x 128 cols
  int m0 = blockIdx.x * 32 + mt * 16;
  int mrow  = m0 + (lane & 15);
  int mload = mrow < nrows ? mrow : (nrows - 1);
  int n = nt * 16 + (lane & 15);

  const v16bf* Blv = (const v16bf*)Wlp;
  const v16bf* Brv = (const v16bf*)Wrp;
  v16bf Bl[4], Br[4];
#pragma unroll
  for (int kc = 0; kc < 4; ++kc) {
    Bl[kc] = Blv[(nt * 4 + kc) * 32 + lane];
    Br[kc] = Brv[(nt * 4 + kc) * 32 + lane];
  }

  const __bf16* pm = meanb + (size_t)mload * DH + ((lane >> 4) * 16);
  const __bf16* ph = hb    + (size_t)mload * DH + ((lane >> 4) * 16);

  v8f acc = {};
#pragma unroll
  for (int kc = 0; kc < 4; ++kc) {
    v16bf am = *(const v16bf*)(pm + kc * 32);
    v16bf ah = *(const v16bf*)(ph + kc * 32);
    acc = __builtin_amdgcn_wmma_f32_16x16x32_bf16(false, am, false, Bl[kc], (short)0, acc, false, false);
    acc = __builtin_amdgcn_wmma_f32_16x16x32_bf16(false, ah, false, Br[kc], (short)0, acc, false, false);
  }

  float s = sc[n];
  float o = sh[n];
  int rbase = m0 + ((lane >> 4) * 8);
#pragma unroll
  for (int r = 0; r < 8; ++r) {
    int m = rbase + r;
    if (m < nrows) {
      float v = acc[r] * s + o;
      hout[(size_t)m * DH + n] = v > 0.0f ? v : 0.0f;
    }
  }
}

// ------------------------------------------ output head: h3 @ Wo + bo (128->64)
__global__ __launch_bounds__(512) void out_gemm_kernel(
    const __bf16* __restrict__ hb, const __bf16* __restrict__ Wop,
    const float* __restrict__ bo, float* __restrict__ out, int nrows) {
  int t = threadIdx.x;
  int wave = t >> 5, lane = t & 31;
  int mt = wave >> 2, nt = wave & 3;            // 4 x 4 tiles -> 64 rows x 64 cols
  int m0 = blockIdx.x * 64 + mt * 16;
  int mrow  = m0 + (lane & 15);
  int mload = mrow < nrows ? mrow : (nrows - 1);
  int n = nt * 16 + (lane & 15);

  const v16bf* Bv = (const v16bf*)Wop;
  v16bf B[4];
#pragma unroll
  for (int kc = 0; kc < 4; ++kc) B[kc] = Bv[(nt * 4 + kc) * 32 + lane];

  const __bf16* ph = hb + (size_t)mload * DH + ((lane >> 4) * 16);

  v8f acc = {};
#pragma unroll
  for (int kc = 0; kc < 4; ++kc) {
    v16bf a = *(const v16bf*)(ph + kc * 32);
    acc = __builtin_amdgcn_wmma_f32_16x16x32_bf16(false, a, false, B[kc], (short)0, acc, false, false);
  }

  float bias = bo[n];
  int rbase = m0 + ((lane >> 4) * 8);
#pragma unroll
  for (int r = 0; r < 8; ++r) {
    int m = rbase + r;
    if (m < nrows) out[(size_t)m * DOUT + n] = acc[r] + bias;
  }
}

// ---------------------------------------------------------------- launch
extern "C" void kernel_launch(void* const* d_in, const int* in_sizes, int n_in,
                              void* d_out, int out_size, void* d_ws, size_t ws_size,
                              hipStream_t stream) {
  const float* x    = (const float*)d_in[0];
  const int*   edge = (const int*)d_in[1];
  int N = in_sizes[0] / DH;
  int E = in_sizes[1] / 2;
  const int* src = edge;
  const int* dst = edge + E;

  const float* Wl[3] = {(const float*)d_in[2], (const float*)d_in[9],  (const float*)d_in[16]};
  const float* bl[3] = {(const float*)d_in[3], (const float*)d_in[10], (const float*)d_in[17]};
  const float* Wr[3] = {(const float*)d_in[4], (const float*)d_in[11], (const float*)d_in[18]};
  const float* g[3]  = {(const float*)d_in[5], (const float*)d_in[12], (const float*)d_in[19]};
  const float* bb[3] = {(const float*)d_in[6], (const float*)d_in[13], (const float*)d_in[20]};
  const float* rm[3] = {(const float*)d_in[7], (const float*)d_in[14], (const float*)d_in[21]};
  const float* rv[3] = {(const float*)d_in[8], (const float*)d_in[15], (const float*)d_in[22]};
  const float* Wo = (const float*)d_in[23];
  const float* bo = (const float*)d_in[24];

  // workspace carve (256B aligned slices)
  char* wsp = (char*)d_ws;
  size_t off = 0;
  auto carve = [&](size_t bytes) -> void* {
    void* p = (void*)(wsp + off);
    off += (bytes + 255) & ~(size_t)255;
    return p;
  };
  float*  agg   = (float*) carve((size_t)N * DH * sizeof(float));
  float*  cnt   = (float*) carve((size_t)N * sizeof(float));
  float*  hA    = (float*) carve((size_t)N * DH * sizeof(float));
  float*  hB    = (float*) carve((size_t)N * DH * sizeof(float));
  __bf16* meanb = (__bf16*)carve((size_t)N * DH * sizeof(__bf16));
  __bf16* hb    = (__bf16*)carve((size_t)N * DH * sizeof(__bf16));
  __bf16* Wlp   = (__bf16*)carve((size_t)DH * DH * sizeof(__bf16));
  __bf16* Wrp   = (__bf16*)carve((size_t)DH * DH * sizeof(__bf16));
  __bf16* Wop   = (__bf16*)carve((size_t)DH * DOUT * sizeof(__bf16));
  float*  sc    = (float*) carve(DH * sizeof(float));
  float*  sh    = (float*) carve(DH * sizeof(float));
  (void)ws_size; (void)n_in; (void)out_size;

  const int thr = 256;
  const long long NH = (long long)N * DH;

  prep_wo_kernel<<<(DH * DOUT + thr - 1) / thr, thr, 0, stream>>>(Wo, Wop);

  const float* hcur = x;
  float* bufs[2] = {hA, hB};
  for (int l = 0; l < 3; ++l) {
    float* hnext = bufs[l & 1];
    prep_layer_kernel<<<(DH * DH + thr - 1) / thr, thr, 0, stream>>>(
        Wl[l], Wr[l], bl[l], g[l], bb[l], rm[l], rv[l], Wlp, Wrp, sc, sh);
    zero_f32_kernel<<<(int)((NH + thr - 1) / thr), thr, 0, stream>>>(agg, NH);
    zero_f32_kernel<<<(N + thr - 1) / thr, thr, 0, stream>>>(cnt, N);
    long long tot = (long long)E * DH;
    scatter_kernel<<<(int)((tot + thr - 1) / thr), thr, 0, stream>>>(hcur, src, dst, agg, cnt, E);
    invcnt_kernel<<<(N + thr - 1) / thr, thr, 0, stream>>>(cnt, N);
    mean_conv_kernel<<<(int)((NH + thr - 1) / thr), thr, 0, stream>>>(agg, cnt, hcur, meanb, hb, N);
    sage_gemm_kernel<<<(N + 31) / 32, 512, 0, stream>>>(meanb, hb, Wlp, Wrp, sc, sh, hnext, N);
    hcur = hnext;
  }

  conv_perm_kernel<<<(int)((NH + thr - 1) / thr), thr, 0, stream>>>(hcur, hb, N);
  out_gemm_kernel<<<(N + 63) / 64, 512, 0, stream>>>(hb, Wop, bo, (float*)d_out, N);
}
